// LinearUnifiedNestedAttention_25812753449183
// MI455X (gfx1250) — compile-verified, matrix operand order checked by reading, not run on hardware
//
#include <hip/hip_runtime.h>
#include <hip/hip_bf16.h>

// ---------------------------------------------------------------------------
// Shapes (compile-time constants from the reference)
// ---------------------------------------------------------------------------
#define B_   16
#define S_   4096
#define P_   64
#define EMB_ 512
#define QKV_ 512
#define H_   8
#define HD_  64
#define PH_  8
#define PHD_ 64

#define MBIG (B_ * S_)              // 65536 rows
#define MSML (B_ * P_)              // 1024 rows
static const size_t NBIG  = (size_t)MBIG * QKV_;   // 33,554,432 elems
static const size_t NSML  = (size_t)MSML * QKV_;   // 524,288 elems
static const size_t WELEM = (size_t)QKV_ * QKV_;   // 262,144 elems

typedef __bf16 bf16;
typedef __attribute__((ext_vector_type(4)))  __bf16 bf16x4;
typedef __attribute__((ext_vector_type(8)))  __bf16 bf16x8;
typedef __attribute__((ext_vector_type(16))) __bf16 v16bf;
typedef __attribute__((ext_vector_type(8)))  float  v8f;
typedef __attribute__((ext_vector_type(4)))  unsigned int u32x4;
typedef __attribute__((ext_vector_type(8)))  int i32x8;
typedef __attribute__((ext_vector_type(4)))  int i32x4;

#if __has_builtin(__builtin_amdgcn_tensor_load_to_lds)
#define USE_TDM 1
#else
#define USE_TDM 0
#endif

// ---------------------------------------------------------------------------
// WMMA helpers (CDNA5: D = A(16x32 bf16) * B(32x16 bf16) + C(16x16 f32))
// ---------------------------------------------------------------------------
__device__ __forceinline__ v8f wmma_bf16(v16bf a, v16bf b, v8f c) {
  return __builtin_amdgcn_wmma_f32_16x16x32_bf16(
      /*neg_a=*/false, a, /*neg_b=*/false, b,
      /*c_mod=*/(short)0, c, /*reuse_a=*/false, /*reuse_b=*/false);
}

__device__ __forceinline__ v8f zero8() {
  v8f z = {0.f, 0.f, 0.f, 0.f, 0.f, 0.f, 0.f, 0.f};
  return z;
}

// Load a 16x32 bf16 fragment (A-layout; identical lane pattern serves as our
// B-layout because we stage B transposed [n][k] in LDS).
// Lane l: row/col index = l&15; lanes 0-15 hold K in {0..7,16..23},
// lanes 16-31 hold K in {8..15,24..31}. Two contiguous 16B runs -> 2x b128.
__device__ __forceinline__ v16bf lds_load_frag(const bf16* base, int stride) {
  int lane = threadIdx.x & 31;
  int r  = lane & 15;
  int kb = (lane < 16) ? 0 : 8;
  const bf16* p = base + r * stride + kb;
  bf16x8 lo = *(const bf16x8*)(p);
  bf16x8 hi = *(const bf16x8*)(p + 16);
  return __builtin_shufflevector(lo, hi,
      0, 1, 2, 3, 4, 5, 6, 7, 8, 9, 10, 11, 12, 13, 14, 15);
}

__device__ __forceinline__ float halfwave_max(float v) {
  v = fmaxf(v, __shfl_xor(v, 1, 32));
  v = fmaxf(v, __shfl_xor(v, 2, 32));
  v = fmaxf(v, __shfl_xor(v, 4, 32));
  v = fmaxf(v, __shfl_xor(v, 8, 32));
  return v;
}
__device__ __forceinline__ float halfwave_sum(float v) {
  v += __shfl_xor(v, 1, 32);
  v += __shfl_xor(v, 2, 32);
  v += __shfl_xor(v, 4, 32);
  v += __shfl_xor(v, 8, 32);
  return v;
}

#if USE_TDM
// ---------------------------------------------------------------------------
// Tensor Data Mover: 2D tile load (128 rows x 32 bf16) global -> LDS.
// D# per CDNA5 ISA:
//   group0: count=1 | lds_addr | global_addr[56:0] | type=2 (bits 127:126)
//   group1: data_size=2B, pad_enable, pad_interval=16 DW, pad_amount=4 DW
//           (reproduces the padded 40-element LDS row stride),
//           tensor_dim0=tile_dim0=32, tensor_dim1=tile_dim1=128,
//           tensor_dim0_stride=512 elems.
// Tracked with TENSORcnt; caller pipelines with s_wait_tensorcnt.
// This toolchain exposes the 6-arg builtin form:
//   (u32x4 g0, i32x8 g1, i32x4 g2, i32x4 g3, i32x8 extra, i32 cpol)
// ---------------------------------------------------------------------------
__device__ __forceinline__ void tdm_load_a_tile(unsigned lds_off, const bf16* gsrc) {
  unsigned long long ga = (unsigned long long)(size_t)(const void*)gsrc;
  u32x4 g0;
  g0[0] = 1u;                                         // count=1, user D#
  g0[1] = lds_off;                                    // lds_addr (bytes)
  g0[2] = (unsigned)ga;                               // global_addr[31:0]
  g0[3] = (unsigned)((ga >> 32) & 0x01FFFFFFu)        // global_addr[56:32]
        | 0x80000000u;                                // type=2 ("image")
  i32x8 g1;
  g1[0] = (int)((1u << 16)      // data_size = 2 bytes
              | (1u << 20)      // pad_enable
              | (3u << 22)      // pad_interval: 16 DWORDs (64B row)
              | (3u << 25));    // pad_amount: 4 DWORDs (16B pad)
  g1[1] = (int)(32u << 16);     // tensor_dim0[15:0]=32 (atomic_addr=0)
  g1[2] = (int)(128u << 16);    // tensor_dim0 hi=0 | tensor_dim1[15:0]=128
  g1[3] = (int)(32u << 16);     // tensor_dim1 hi=0 | tile_dim0=32
  g1[4] = (int)128;             // tile_dim1=128, tile_dim2=0
  g1[5] = (int)512;             // tensor_dim0_stride=512 elems
  g1[6] = 0;
  g1[7] = 0;
  i32x4 gz4 = {0, 0, 0, 0};                 // groups 2/3 unused (2D tensor)
  i32x8 gz8 = {0, 0, 0, 0, 0, 0, 0, 0};
  __builtin_amdgcn_tensor_load_to_lds(g0, g1, gz4, gz4, gz8, 0);
}
#endif

// ---------------------------------------------------------------------------
// fp32 -> bf16 conversion (vectorized x4)
// ---------------------------------------------------------------------------
__global__ void __launch_bounds__(256)
cvt_f32_bf16_kernel(const float* __restrict__ in, bf16* __restrict__ out, int n4) {
  int i = blockIdx.x * 256 + threadIdx.x;
  if (i < n4) {
    float4 v = ((const float4*)in)[i];
    bf16x4 o;
    o[0] = (bf16)v.x; o[1] = (bf16)v.y; o[2] = (bf16)v.z; o[3] = (bf16)v.w;
    ((bf16x4*)out)[i] = o;
  }
}

// ---------------------------------------------------------------------------
// GEMM: C[M,512] = (A[M,512] @ W[512,512] + bias) * scale
// Block tile 128x64, 8 waves (4x2), each wave 32x32 (2x2 WMMA tiles), BK=32.
// A tiles streamed by the TDM into a double-buffered padded LDS region
// (wave 0 issues; s_wait_tensorcnt(1) keeps the next tile in flight during
// compute). W staged transposed [n][k] by the SIMDs. Stride 40 bf16 rows.
// ---------------------------------------------------------------------------
__global__ void __launch_bounds__(256)
gemm_bias_kernel(const bf16* __restrict__ A, const bf16* __restrict__ W,
                 const float* __restrict__ bias, float scale,
                 bf16* __restrict__ outb, float* __restrict__ outf) {
  __shared__ __attribute__((aligned(16))) bf16 Alds[2][128][40];
  __shared__ __attribute__((aligned(16))) bf16 Wlds[64][40];

  const int n0g = blockIdx.x * 64;
  const int m0g = blockIdx.y * 128;
  const int tid  = threadIdx.x;
  const int w    = tid >> 5;
  const int lane = tid & 31;
  const int wm   = (w >> 1) * 32;       // wave M offset within block
  const int wn   = (w & 1) * 32;        // wave N offset within block
  const int r    = lane & 15;
  const int half = lane >> 4;
  const bool w0  = (w == 0);

  v8f c[2][2];
#pragma unroll
  for (int i = 0; i < 2; ++i)
#pragma unroll
    for (int j = 0; j < 2; ++j) c[i][j] = zero8();

  const bf16* Abase = A + (size_t)m0g * 512;

#if USE_TDM
  const unsigned lds_off0 = (unsigned)(size_t)(void*)&Alds[0][0][0];
  const unsigned lds_off1 = (unsigned)(size_t)(void*)&Alds[1][0][0];
  if (w0) tdm_load_a_tile(lds_off0, Abase);           // prologue: tile kb=0
#endif

  for (int kb = 0; kb < 16; ++kb) {
    const int k0  = kb * 32;
    const int cur = kb & 1;
    __syncthreads();   // prev compute done: Wlds + A buffer (cur^1) reusable

#if USE_TDM
    if (w0 && kb + 1 < 16)  // issue next A tile into the other buffer
      tdm_load_a_tile(cur ? lds_off0 : lds_off1, Abase + k0 + 32);
#else
    // Fallback: manual A staging (128x32 bf16 = 512 x bf16x8; 2 per thread).
#pragma unroll
    for (int it = 0; it < 2; ++it) {
      int idx = tid + it * 256;
      int row = idx >> 2, cb = idx & 3;
      bf16x8 va = *(const bf16x8*)(Abase + (size_t)row * 512 + k0 + cb * 8);
      *(bf16x8*)&Alds[cur][row][cb * 8] = va;
    }
#endif

    // Stage W tile transposed: rows k0..k0+31, cols n0g..n0g+63.
    {
      int row = tid >> 3, cb = tid & 7;
      bf16x8 vw = *(const bf16x8*)(W + (size_t)(k0 + row) * 512 + n0g + cb * 8);
#pragma unroll
      for (int j = 0; j < 8; ++j) Wlds[cb * 8 + j][row] = vw[j];
    }
    if (k0 + 32 < 512)  // hint next W tile into cache
      __builtin_prefetch(W + (size_t)(k0 + 32 + (tid >> 3)) * 512 + n0g, 0, 1);

#if USE_TDM
    if (w0) {
      if (kb + 1 < 16) __builtin_amdgcn_s_wait_tensorcnt(1);  // cur done, next in flight
      else             __builtin_amdgcn_s_wait_tensorcnt(0);
    }
#endif
    __syncthreads();

    v16bf af0 = lds_load_frag(&Alds[cur][wm][0], 40);
    v16bf af1 = lds_load_frag(&Alds[cur][wm + 16][0], 40);
    v16bf bf0 = lds_load_frag(&Wlds[wn][0], 40);
    v16bf bf1 = lds_load_frag(&Wlds[wn + 16][0], 40);
    c[0][0] = wmma_bf16(af0, bf0, c[0][0]);
    c[0][1] = wmma_bf16(af0, bf1, c[0][1]);
    c[1][0] = wmma_bf16(af1, bf0, c[1][0]);
    c[1][1] = wmma_bf16(af1, bf1, c[1][1]);
  }

  // Epilogue: bias + scale, write bf16 and/or fp32.
#pragma unroll
  for (int j = 0; j < 2; ++j) {
    int n = n0g + wn + j * 16 + r;
    float bval = bias[n];
#pragma unroll
    for (int i = 0; i < 2; ++i) {
#pragma unroll
      for (int rr = 0; rr < 8; ++rr) {
        int m = m0g + wm + i * 16 + rr + 8 * half;
        float v = (c[i][j][rr] + bval) * scale;
        size_t off = (size_t)m * 512 + n;
        if (outb) outb[off] = (bf16)v;
        if (outf) outf[off] = v;
      }
    }
  }
}

// ---------------------------------------------------------------------------
// Pack stage: per (b, ph) head, pquery (P=64) attends over context (S=4096).
// Flash-softmax streamed in chunks of 64 along S. 4 waves x 16 rows of P.
// pc[b, p, ph*64 + d] written in bf16.
// ---------------------------------------------------------------------------
__global__ void __launch_bounds__(128)
pack_attn_kernel(const bf16* __restrict__ pq, const bf16* __restrict__ pk,
                 const bf16* __restrict__ pv, bf16* __restrict__ pc_bf) {
  __shared__ __attribute__((aligned(16))) bf16 pk_lds[64][72];   // [s][d]
  __shared__ __attribute__((aligned(16))) bf16 pvT_lds[64][72];  // [d][s]
  __shared__ __attribute__((aligned(16))) bf16 pexp_lds[4][16][72];

  const int b  = blockIdx.x >> 3;
  const int ph = blockIdx.x & 7;
  const int tid  = threadIdx.x;
  const int w    = tid >> 5;
  const int lane = tid & 31;
  const int r    = lane & 15;
  const int half = lane >> 4;
  const int kb   = half ? 8 : 0;

  // Preload pq strip (rows p = w*16 .. +15, K = PHD=64) directly in A-layout.
  v16bf aq[2];
  const bf16* pqrow = pq + (size_t)(b * 64 + w * 16 + r) * 512 + ph * 64;
#pragma unroll
  for (int kk = 0; kk < 2; ++kk)
#pragma unroll
    for (int i = 0; i < 16; ++i) {
      int k = kk * 32 + (i < 8 ? kb + i : 16 + kb + (i - 8));
      aq[kk][i] = pqrow[k];
    }

  v8f acc[4];
  float m_run[8], l_run[8];
#pragma unroll
  for (int t = 0; t < 4; ++t) acc[t] = zero8();
#pragma unroll
  for (int rr = 0; rr < 8; ++rr) { m_run[rr] = -1e30f; l_run[rr] = 0.f; }

  for (int sc = 0; sc < S_; sc += 64) {
    __syncthreads();
    // Stage pk chunk [s][d] and pv chunk transposed [d][s].
#pragma unroll
    for (int it = 0; it < 4; ++it) {
      int idx = tid + it * 128;        // 0..511
      int srow = idx >> 3, cb = idx & 7;
      size_t g = (size_t)(b * S_ + sc + srow) * 512 + ph * 64 + cb * 8;
      bf16x8 vk = *(const bf16x8*)(pk + g);
      *(bf16x8*)&pk_lds[srow][cb * 8] = vk;
      bf16x8 vv = *(const bf16x8*)(pv + g);
#pragma unroll
      for (int j = 0; j < 8; ++j) pvT_lds[cb * 8 + j][srow] = vv[j];
    }
    __syncthreads();

    // scores(16 x 64) = pq_strip @ pk^T  (N = s within chunk, K = d)
    v8f s4[4];
#pragma unroll
    for (int t = 0; t < 4; ++t) {
      v8f cfr = zero8();
#pragma unroll
      for (int kk = 0; kk < 2; ++kk) {
        v16bf bfr = lds_load_frag(&pk_lds[t * 16][kk * 32], 72);
        cfr = wmma_bf16(aq[kk], bfr, cfr);
      }
      s4[t] = cfr;
    }

    // Online softmax update (row stats per C-layout VGPR / half-wave).
#pragma unroll
    for (int rr = 0; rr < 8; ++rr) {
      float mv = fmaxf(fmaxf(s4[0][rr], s4[1][rr]), fmaxf(s4[2][rr], s4[3][rr]));
      mv = halfwave_max(mv);
      float mnew = fmaxf(m_run[rr], mv);
      float corr = __expf(m_run[rr] - mnew);
      m_run[rr] = mnew;
      float lsum = 0.f;
#pragma unroll
      for (int t = 0; t < 4; ++t) {
        float e = __expf(s4[t][rr] - mnew);
        s4[t][rr] = e;
        lsum += e;
      }
      lsum = halfwave_sum(lsum);
      l_run[rr] = l_run[rr] * corr + lsum;
#pragma unroll
      for (int t = 0; t < 4; ++t) acc[t][rr] *= corr;
    }

    // C-layout -> A-layout via per-wave LDS strip (bf16).
#pragma unroll
    for (int rr = 0; rr < 8; ++rr)
#pragma unroll
      for (int t = 0; t < 4; ++t)
        pexp_lds[w][rr + 8 * half][t * 16 + r] = (bf16)s4[t][rr];
    __syncthreads();

    // acc += exp(P) @ pv_chunk   (N = d, K = s within chunk)
#pragma unroll
    for (int t = 0; t < 4; ++t) {
#pragma unroll
      for (int kk = 0; kk < 2; ++kk) {
        v16bf af  = lds_load_frag(&pexp_lds[w][0][kk * 32], 72);
        v16bf bfr = lds_load_frag(&pvT_lds[t * 16][kk * 32], 72);
        acc[t] = wmma_bf16(af, bfr, acc[t]);
      }
    }
  }

  // Normalize and store pc in bf16.
#pragma unroll
  for (int rr = 0; rr < 8; ++rr) {
    float inv = 1.f / l_run[rr];
    int prow = w * 16 + rr + 8 * half;
#pragma unroll
    for (int t = 0; t < 4; ++t) {
      size_t off = (size_t)(b * 64 + prow) * 512 + ph * 64 + t * 16 + r;
      pc_bf[off] = (bf16)(acc[t][rr] * inv);
    }
  }
}

// ---------------------------------------------------------------------------
// Unpack stage: per (b, h, 64-row S tile): q attends over p_context (P=64).
// Exact softmax over P. Writes aw_t (S, B*H, P) fp32 and attn bf16.
// ---------------------------------------------------------------------------
__global__ void __launch_bounds__(128)
unpack_attn_kernel(const bf16* __restrict__ qm, const bf16* __restrict__ km,
                   const bf16* __restrict__ vm, float* __restrict__ aw_out,
                   bf16* __restrict__ attn_bf) {
  __shared__ __attribute__((aligned(16))) bf16 k_lds[64][72];   // [p][d]
  __shared__ __attribute__((aligned(16))) bf16 vT_lds[64][72];  // [d][p]
  __shared__ __attribute__((aligned(16))) bf16 pexp_lds[4][16][72];

  const int bh = blockIdx.x;
  const int b  = bh >> 3;
  const int h  = bh & 7;
  const int s0 = blockIdx.y * 64;
  const int tid  = threadIdx.x;
  const int w    = tid >> 5;
  const int lane = tid & 31;
  const int r    = lane & 15;
  const int half = lane >> 4;
  const int kb   = half ? 8 : 0;

  // Stage k [p][d] and v transposed [d][p].
#pragma unroll
  for (int it = 0; it < 4; ++it) {
    int idx = tid + it * 128;
    int prow = idx >> 3, cb = idx & 7;
    size_t g = (size_t)(b * 64 + prow) * 512 + h * 64 + cb * 8;
    bf16x8 vk = *(const bf16x8*)(km + g);
    *(bf16x8*)&k_lds[prow][cb * 8] = vk;
    bf16x8 vv = *(const bf16x8*)(vm + g);
#pragma unroll
    for (int j = 0; j < 8; ++j) vT_lds[cb * 8 + j][prow] = vv[j];
  }

  // q strip in A-layout straight from global.
  v16bf aq[2];
  const bf16* qrow = qm + (size_t)(b * S_ + s0 + w * 16 + r) * 512 + h * 64;
#pragma unroll
  for (int kk = 0; kk < 2; ++kk)
#pragma unroll
    for (int i = 0; i < 16; ++i) {
      int k = kk * 32 + (i < 8 ? kb + i : 16 + kb + (i - 8));
      aq[kk][i] = qrow[k];
    }
  __syncthreads();

  // scores(16 x 64) = q_strip @ k^T  (N = p, K = d)
  v8f s4[4];
#pragma unroll
  for (int t = 0; t < 4; ++t) {
    v8f cfr = zero8();
#pragma unroll
    for (int kk = 0; kk < 2; ++kk) {
      v16bf bfr = lds_load_frag(&k_lds[t * 16][kk * 32], 72);
      cfr = wmma_bf16(aq[kk], bfr, cfr);
    }
    s4[t] = cfr;
  }

  // Exact softmax over N = P = 64; write aw (fp32) and staged bf16.
#pragma unroll
  for (int rr = 0; rr < 8; ++rr) {
    float mv = fmaxf(fmaxf(s4[0][rr], s4[1][rr]), fmaxf(s4[2][rr], s4[3][rr]));
    mv = halfwave_max(mv);
    float lsum = 0.f;
#pragma unroll
    for (int t = 0; t < 4; ++t) {
      float e = __expf(s4[t][rr] - mv);
      s4[t][rr] = e;
      lsum += e;
    }
    lsum = halfwave_sum(lsum);
    float inv = 1.f / lsum;
    int s = s0 + w * 16 + rr + 8 * half;
#pragma unroll
    for (int t = 0; t < 4; ++t) {
      float aw = s4[t][rr] * inv;
      s4[t][rr] = aw;
      aw_out[(size_t)s * (B_ * H_ * P_) + (size_t)bh * P_ + t * 16 + r] = aw;
      pexp_lds[w][rr + 8 * half][t * 16 + r] = (bf16)aw;
    }
  }
  __syncthreads();

  // attn(16 x 64) = aw @ v  (N = d, K = p)
  v8f acc[4];
#pragma unroll
  for (int t = 0; t < 4; ++t) {
    v8f cfr = zero8();
#pragma unroll
    for (int kk = 0; kk < 2; ++kk) {
      v16bf af  = lds_load_frag(&pexp_lds[w][0][kk * 32], 72);
      v16bf bfr = lds_load_frag(&vT_lds[t * 16][kk * 32], 72);
      cfr = wmma_bf16(af, bfr, cfr);
    }
    acc[t] = cfr;
  }

#pragma unroll
  for (int rr = 0; rr < 8; ++rr) {
    int s = s0 + w * 16 + rr + 8 * half;
#pragma unroll
    for (int t = 0; t < 4; ++t) {
      size_t off = (size_t)(b * S_ + s) * 512 + h * 64 + t * 16 + r;
      attn_bf[off] = (bf16)acc[t][rr];
    }
  }
}

// ---------------------------------------------------------------------------
// Host-side orchestration
// ---------------------------------------------------------------------------
static inline void launch_cvt(const float* in, bf16* out, size_t n, hipStream_t s) {
  int n4 = (int)(n / 4);
  cvt_f32_bf16_kernel<<<(n4 + 255) / 256, 256, 0, s>>>(in, out, n4);
}

static inline void launch_gemm(const bf16* A, const bf16* W, const float* bias,
                               float scale, bf16* outb, float* outf, int M,
                               hipStream_t s) {
  dim3 grid(QKV_ / 64, M / 128);
  gemm_bias_kernel<<<grid, 256, 0, s>>>(A, W, bias, scale, outb, outf);
}

extern "C" void kernel_launch(void* const* d_in, const int* in_sizes, int n_in,
                              void* d_out, int out_size, void* d_ws, size_t ws_size,
                              hipStream_t stream) {
  (void)in_sizes; (void)n_in; (void)out_size; (void)ws_size;
  // Inputs (setup_inputs dict order)
  const float* query   = (const float*)d_in[0];
  const float* pquery  = (const float*)d_in[1];
  const float* context = (const float*)d_in[2];
  const float* Wi  = (const float*)d_in[3];  const float* bi  = (const float*)d_in[4];
  const float* Wip = (const float*)d_in[5];  const float* bip = (const float*)d_in[6];
  const float* Wic = (const float*)d_in[7];  const float* bic = (const float*)d_in[8];
  const float* Wq  = (const float*)d_in[9];  const float* bq  = (const float*)d_in[10];
  const float* Wpq = (const float*)d_in[11]; const float* bpq = (const float*)d_in[12];
  const float* Wk  = (const float*)d_in[13]; const float* bk  = (const float*)d_in[14];
  const float* Wpk = (const float*)d_in[15]; const float* bpk = (const float*)d_in[16];
  const float* Wv  = (const float*)d_in[17]; const float* bv  = (const float*)d_in[18];
  const float* Wpv = (const float*)d_in[19]; const float* bpv = (const float*)d_in[20];
  const float* Wo  = (const float*)d_in[21]; const float* bo  = (const float*)d_in[22];
  const float* Wop = (const float*)d_in[23]; const float* bop = (const float*)d_in[24];

  // Outputs: attn (B,S,EMB) | pc_out (B,P,EMB) | aw_t (S, B*H, P), all fp32.
  float* out_attn = (float*)d_out;
  float* out_pc   = out_attn + NBIG;
  float* out_aw   = out_pc + NSML;

  // Workspace cursor allocation (256B aligned).
  char* wsp = (char*)d_ws;
  size_t off = 0;
  auto take = [&](size_t bytes) -> void* {
    void* p = wsp + off;
    off += (bytes + 255) & ~(size_t)255;
    return p;
  };
  bf16* bufA = (bf16*)take(NBIG * 2);   // query_bf -> q (reuse)
  bf16* bufB = (bf16*)take(NBIG * 2);   // context_bf -> attn (reuse)
  bf16* bufC = (bf16*)take(NBIG * 2);   // q_in
  bf16* bufD = (bf16*)take(NBIG * 2);   // c_in
  bf16* bufE = (bf16*)take(NBIG * 2);   // pk
  bf16* bufF = (bf16*)take(NBIG * 2);   // pv
  bf16* pquery_bf = (bf16*)take(NSML * 2);
  bf16* pq_in_bf  = (bf16*)take(NSML * 2);
  bf16* pqs_bf    = (bf16*)take(NSML * 2);   // scaled pq
  bf16* pc_bf     = (bf16*)take(NSML * 2);
  bf16* k_bf      = (bf16*)take(NSML * 2);
  bf16* v_bf      = (bf16*)take(NSML * 2);
  bf16* wbf[11];
  for (int i = 0; i < 11; ++i) wbf[i] = (bf16*)take(WELEM * 2);
  // wbf index: 0 Wi, 1 Wip, 2 Wic, 3 Wq, 4 Wpq, 5 Wk, 6 Wpk, 7 Wv, 8 Wpv, 9 Wo, 10 Wop

  // --- Convert activations and weights to bf16 ---
  launch_cvt(query,   bufA,      NBIG, stream);
  launch_cvt(pquery,  pquery_bf, NSML, stream);
  launch_cvt(context, bufB,      NBIG, stream);
  const float* wsrc[11] = {Wi, Wip, Wic, Wq, Wpq, Wk, Wpk, Wv, Wpv, Wo, Wop};
  for (int i = 0; i < 11; ++i) launch_cvt(wsrc[i], wbf[i], WELEM, stream);

  const float inv8 = 0.125f;  // HD^-0.5 == PHD^-0.5 == 1/8

  // --- Input projections ---
  launch_gemm(bufA,      wbf[0], bi,  1.f, bufC,     nullptr, MBIG, stream); // q_in
  launch_gemm(pquery_bf, wbf[1], bip, 1.f, pq_in_bf, nullptr, MSML, stream); // pq_in
  launch_gemm(bufB,      wbf[2], bic, 1.f, bufD,     nullptr, MBIG, stream); // c_in

  // --- Pack stage projections + attention ---
  launch_gemm(pq_in_bf, wbf[4], bpq, inv8, pqs_bf, nullptr, MSML, stream);   // pq (scaled)
  launch_gemm(bufD,     wbf[6], bpk, 1.f,  bufE,   nullptr, MBIG, stream);   // pk
  launch_gemm(bufD,     wbf[8], bpv, 1.f,  bufF,   nullptr, MBIG, stream);   // pv
  pack_attn_kernel<<<B_ * PH_, 128, 0, stream>>>(pqs_bf, bufE, bufF, pc_bf);

  // --- Unpack stage projections + attention ---
  launch_gemm(bufC,  wbf[3], bq, inv8, bufA, nullptr, MBIG, stream);         // q (scaled)
  launch_gemm(pc_bf, wbf[5], bk, 1.f,  k_bf, nullptr, MSML, stream);         // k
  launch_gemm(pc_bf, wbf[7], bv, 1.f,  v_bf, nullptr, MSML, stream);         // v
  {
    dim3 grid(B_ * H_, S_ / 64);
    unpack_attn_kernel<<<grid, 128, 0, stream>>>(bufA, k_bf, v_bf, out_aw, bufB);
  }

  // --- Output projections (fp32 epilogue into d_out) ---
  launch_gemm(bufB,  wbf[9],  bo,  1.f, nullptr, out_attn, MBIG, stream);
  launch_gemm(pc_bf, wbf[10], bop, 1.f, nullptr, out_pc,   MSML, stream);
}
